// RecurrentModel_9337258901774
// MI455X (gfx1250) — matrix operand order repaired and checked by reading, not compiled, not run on hardware
//
#include <hip/hip_runtime.h>
#include <cstdint>
#include <cstddef>

// Problem constants (match reference): B=64, T=512, I=512, H=1024
#define BB 64
#define TT 512
#define II 512
#define HH 1024
#define G3 (3 * HH)

typedef __attribute__((ext_vector_type(16))) __bf16        v16bf;
typedef __attribute__((ext_vector_type(8)))  float         v8f;
typedef __attribute__((ext_vector_type(4)))  unsigned int  u32x4;
typedef __attribute__((ext_vector_type(4)))  float         f32x4;

union Frag16 { v16bf v; u32x4 q[2]; };

// Load a 32-byte B fragment (ISA layout: elems 0-7 at k, elems 8-15 at k+16)
#define LDB(F, p)                                   \
  do {                                              \
    (F).q[0] = *(const u32x4*)(p);                  \
    (F).q[1] = *(const u32x4*)((p) + 16);           \
  } while (0)

// Load a 32-byte A fragment (ISA layout: elems 0-7 at k, elems 8-15 at k+16)
#define LDA(F, p)                                   \
  do {                                              \
    (F).q[0] = *(const u32x4*)(p);                  \
    (F).q[1] = *(const u32x4*)((p) + 16);           \
  } while (0)

#define WMMA(acc, A, Bf)                                          \
  acc = __builtin_amdgcn_wmma_f32_16x16x32_bf16(                  \
      false, (A).v, false, (Bf).v, (short)0, acc, false, false)

// f32 -> bf16 round-to-nearest-even
__device__ __forceinline__ unsigned short f2bf(float f) {
  unsigned int u = __float_as_uint(f);
  u += 0x7FFFu + ((u >> 16) & 1u);
  return (unsigned short)(u >> 16);
}

__device__ __forceinline__ float sigmoidf_(float x) {
  return 1.0f / (1.0f + __expf(-x));
}

__device__ __forceinline__ v8f splat8(float v) {
  v8f s = {v, v, v, v, v, v, v, v};
  return s;
}

// ---------------------------------------------------------------------------
// Kernel 1: one-time f32 -> bf16 conversion of W_ih [3H,I] and W_hh [3H,H]
// into workspace (9 MB total). Stays L2-resident (192 MB) for the whole scan.
// ---------------------------------------------------------------------------
__global__ void cvt_weights_kernel(const float* __restrict__ Wih,
                                   const float* __restrict__ Whh,
                                   unsigned short* __restrict__ WihB,
                                   unsigned short* __restrict__ WhhB) {
  const int n1 = G3 * II;
  const int n2 = G3 * HH;
  for (int i = blockIdx.x * blockDim.x + threadIdx.x; i < n1 + n2;
       i += gridDim.x * blockDim.x) {
    if (i < n1) WihB[i] = f2bf(Wih[i]);
    else        WhhB[i - n1] = f2bf(Whh[i - n1]);
  }
}

// ---------------------------------------------------------------------------
// Kernel 2: persistent GRU scan. One workgroup per 16-row batch slice
// (recurrence is independent per batch row -> no inter-WG sync needed).
// 32 waves; each wave owns 32 H-columns (2 column blocks x 3 gates).
// Register plan: 8 live v8f accumulators (r/z fused across x- and h-GEMMs;
// only the n-gate keeps gx and gh separate, as GRU semantics require).
// ---------------------------------------------------------------------------
__launch_bounds__(1024, 1)
__global__ void gru_scan_kernel(const float* __restrict__ x,            // [B,T,I]
                                const unsigned short* __restrict__ WihB,// [3H,I] bf16
                                const unsigned short* __restrict__ WhhB,// [3H,H] bf16
                                const float* __restrict__ b_ih,         // [3H]
                                const float* __restrict__ b_hh,         // [3H]
                                float* __restrict__ out) {              // [B,H]
  extern __shared__ char smem[];
  unsigned short* xbf = (unsigned short*)smem;      // 16 x II  bf16 (16 KB)
  unsigned short* hbf = xbf + 16 * II;              // 16 x HH  bf16 (32 KB)
  float*          hf  = (float*)(hbf + 16 * HH);    // 16 x HH  f32  (64 KB)

  const int tid  = threadIdx.x;
  const int lane = tid & 31;
  const int w    = tid >> 5;            // wave id 0..31
  const int b0   = blockIdx.x * 16;     // batch slice base row

  // Zero-init recurrent state h0 = 0
  for (int i = tid; i < 16 * HH; i += 1024) { hf[i] = 0.0f; hbf[i] = 0; }

  // Per-lane fragment geometry (wave32 WMMA layouts, CDNA5 ISA 7.12.2)
  const int nloc  = lane & 15;          // tile column this lane owns (B/C/D)
  const int khA   = (lane >> 4) * 8;    // A-frag K sub-offset
  const int khB   = (lane >> 4) * 8;    // B-frag K sub-offset
  const int mbase = (lane >> 4) * 8;    // C/D row base for this lane
  const int jc0   = w * 32 + nloc;      // H-column, block 0 (block 1 = +16)

  // Per-lane biases (combined for r/z; separate for n per GRU semantics)
  const float br0  = b_ih[0 * HH + jc0]      + b_hh[0 * HH + jc0];
  const float br1  = b_ih[0 * HH + jc0 + 16] + b_hh[0 * HH + jc0 + 16];
  const float bz0  = b_ih[1 * HH + jc0]      + b_hh[1 * HH + jc0];
  const float bz1  = b_ih[1 * HH + jc0 + 16] + b_hh[1 * HH + jc0 + 16];
  const float bxn0 = b_ih[2 * HH + jc0];
  const float bxn1 = b_ih[2 * HH + jc0 + 16];
  const float bhn0 = b_hh[2 * HH + jc0];
  const float bhn1 = b_hh[2 * HH + jc0 + 16];

  // Per-lane weight bases; gate (g*H*K) and column-block (16*K) offsets are
  // compile-time constants that fold into the 24-bit global-load immediates.
  const unsigned short* wihL = WihB + (size_t)jc0 * II + khB;
  const unsigned short* whhL = WhhB + (size_t)jc0 * HH + khB;

  // A-fragment LDS bases
  const unsigned short* xA = &xbf[nloc * II + khA];
  const unsigned short* hA = &hbf[nloc * HH + khA];

  // Combine-phase LDS index base: row mbase, column jc0
  const int ibase = mbase * HH + jc0;

  // x staging assignment: thread -> (row sm, 8 consecutive i at si)
  const int sm = tid >> 6;
  const int si = (tid & 63) * 8;
  const float* xrow = x + ((size_t)(b0 + sm) * TT) * II + si;

  for (int t = 0; t < TT; ++t) {
    // ---- stage x_t into LDS as bf16 (prev step finished reading xbf) ----
    {
      const float* xp = xrow + (size_t)t * II;
      f32x4 a0 = *(const f32x4*)(xp);
      f32x4 a1 = *(const f32x4*)(xp + 4);
      unsigned short* dst = &xbf[sm * II + si];
      dst[0] = f2bf(a0[0]); dst[1] = f2bf(a0[1]);
      dst[2] = f2bf(a0[2]); dst[3] = f2bf(a0[3]);
      dst[4] = f2bf(a1[0]); dst[5] = f2bf(a1[1]);
      dst[6] = f2bf(a1[2]); dst[7] = f2bf(a1[3]);
      if (t + 1 < TT) __builtin_prefetch(xp + II, 0, 1);  // global_prefetch next x
    }
    __syncthreads();   // xbf ready; h_{t-1} writes visible to all waves

    // Accumulators, biases pre-splatted (C-matrix of the first WMMA).
    v8f ar0 = splat8(br0),  ar1 = splat8(br1);     // r: fused x+h
    v8f az0 = splat8(bz0),  az1 = splat8(bz1);     // z: fused x+h
    v8f axn0 = splat8(bxn0), axn1 = splat8(bxn1);  // n: x part (+b_ih_n)
    v8f ahn0, ahn1;                                // n: h part (+b_hh_n)

    // ---- x contribution: K = II = 512 ----
#pragma unroll 1
    for (int k0 = 0; k0 < II; k0 += 32) {
      Frag16 A, B0, B1;
      LDA(A, xA + k0);
      const unsigned short* bp = wihL + k0;
      LDB(B0, bp);                 LDB(B1, bp + 16 * II);
      WMMA(ar0, A, B0);            WMMA(ar1, A, B1);
      LDB(B0, bp + 1 * HH * II);   LDB(B1, bp + 1 * HH * II + 16 * II);
      WMMA(az0, A, B0);            WMMA(az1, A, B1);
      LDB(B0, bp + 2 * HH * II);   LDB(B1, bp + 2 * HH * II + 16 * II);
      WMMA(axn0, A, B0);           WMMA(axn1, A, B1);
    }

    ahn0 = splat8(bhn0); ahn1 = splat8(bhn1);

    // ---- h contribution: K = HH = 1024 ----
#pragma unroll 1
    for (int k0 = 0; k0 < HH; k0 += 32) {
      Frag16 A, B0, B1;
      LDA(A, hA + k0);
      const unsigned short* bp = whhL + k0;
      LDB(B0, bp);                 LDB(B1, bp + 16 * HH);
      WMMA(ar0, A, B0);            WMMA(ar1, A, B1);
      LDB(B0, bp + 1 * HH * HH);   LDB(B1, bp + 1 * HH * HH + 16 * HH);
      WMMA(az0, A, B0);            WMMA(az1, A, B1);
      LDB(B0, bp + 2 * HH * HH);   LDB(B1, bp + 2 * HH * HH + 16 * HH);
      WMMA(ahn0, A, B0);           WMMA(ahn1, A, B1);
    }

    __syncthreads();   // all waves done reading hbf for this step

    // ---- gate combine (f32); each wave rewrites only its own 32 H-cols ----
#pragma unroll
    for (int e = 0; e < 8; ++e) {
      const int idx = ibase + e * HH;
      // column block 0
      {
        float r  = sigmoidf_(ar0[e]);
        float z  = sigmoidf_(az0[e]);
        float n_ = tanhf(axn0[e] + r * ahn0[e]);
        float hold = hf[idx];
        float hnew = (1.0f - z) * n_ + z * hold;
        hf[idx]  = hnew;
        hbf[idx] = f2bf(hnew);
      }
      // column block 1
      {
        float r  = sigmoidf_(ar1[e]);
        float z  = sigmoidf_(az1[e]);
        float n_ = tanhf(axn1[e] + r * ahn1[e]);
        float hold = hf[idx + 16];
        float hnew = (1.0f - z) * n_ + z * hold;
        hf[idx + 16]  = hnew;
        hbf[idx + 16] = f2bf(hnew);
      }
    }
    // next iteration's __syncthreads() orders these writes vs. next reads
  }

  __syncthreads();
  // ---- epilogue: ReLU(h_T) -> out ----
  for (int i = tid; i < 16 * HH; i += 1024) {
    const int m = i >> 10;          // HH == 1024
    const int k = i & (HH - 1);
    float v = hf[i];
    out[(size_t)(b0 + m) * HH + k] = v > 0.0f ? v : 0.0f;
  }
}

// ---------------------------------------------------------------------------
// Launcher
// ---------------------------------------------------------------------------
extern "C" void kernel_launch(void* const* d_in, const int* in_sizes, int n_in,
                              void* d_out, int out_size, void* d_ws, size_t ws_size,
                              hipStream_t stream) {
  const float* x    = (const float*)d_in[0];   // [B,T,I]
  const float* Wih  = (const float*)d_in[1];   // [3H,I]
  const float* Whh  = (const float*)d_in[2];   // [3H,H]
  const float* bih  = (const float*)d_in[3];   // [3H]
  const float* bhh  = (const float*)d_in[4];   // [3H]
  float* out = (float*)d_out;                  // [B,H]

  unsigned short* WihB = (unsigned short*)d_ws;               // 3 MB
  unsigned short* WhhB = WihB + (size_t)G3 * II;              // 6 MB
  (void)in_sizes; (void)n_in; (void)out_size; (void)ws_size;

  cvt_weights_kernel<<<512, 256, 0, stream>>>(Wih, Whh, WihB, WhhB);

  const size_t ldsBytes = (size_t)(16 * II + 16 * HH) * sizeof(unsigned short)
                        + (size_t)(16 * HH) * sizeof(float);   // 112 KB / WGP
  gru_scan_kernel<<<BB / 16, 1024, ldsBytes, stream>>>(x, WihB, WhhB, bih, bhh, out);
}
